// Attention_90074054132266
// MI455X (gfx1250) — compile-verified, matrix operand order checked by reading, not run on hardware
//
#include <hip/hip_runtime.h>
#include <hip/hip_bf16.h>

typedef __attribute__((ext_vector_type(16))) _Float16 v16h;
typedef __attribute__((ext_vector_type(8)))  _Float16 v8h;
typedef __attribute__((ext_vector_type(4)))  _Float16 v4h;
typedef __attribute__((ext_vector_type(8)))  float    v8f;

#define DIM_   512
#define HEADS_ 8
#define HD_    64
#define B_     2
#define S_     4096
#define ROWS_  (B_ * S_)   /* 8192 */

// CDNA5 16-bit A/B fragment element -> K index (CDNA5 ISA):
// lane group g = lane>>4, element e in v16h:  K = (e&8)*2 + (e&7) + g*8
__device__ __forceinline__ int kmap(int e, int g) {
    return ((e & 8) << 1) + (e & 7) + (g << 3);
}

__device__ __forceinline__ v8f wmma_f32_16x16x32_f16(v16h a, v16h b, v8f c) {
    return __builtin_amdgcn_wmma_f32_16x16x32_f16(false, a, false, b, (short)0, c, false, false);
}

// Load a 16x32 fragment row from a contiguous 64-half global row:
// halves 0..7 <- row[base + g*8 ..], halves 8..15 <- row[base + 16 + g*8 ..]
__device__ __forceinline__ v16h load_frag_global(const _Float16* row, int kk, int g) {
    v16h f;
    v8h lo = *(const v8h*)(row + kk * 32 + g * 8);
    v8h hi = *(const v8h*)(row + kk * 32 + 16 + g * 8);
    #pragma unroll
    for (int e = 0; e < 8; ++e) { f[e] = lo[e]; f[e + 8] = hi[e]; }
    return f;
}

// -ln(10000)/32  (RoPE frequency exponent per column, for __expf)
#define ROPE_C (-0.28782313662425f)

// ---------------------------------------------------------------------------
// Kernel 1: QKV projection + bias + RoPE, f32 in -> f16 out, head-major.
// Tile: M=64 tokens x N=64 (exactly one head of one of {Q,K,V}).
// Weight tile staged TRANSPOSED in LDS so B-fragment gathers are contiguous.
// ---------------------------------------------------------------------------
__global__ __launch_bounds__(128)
void qkv_rope_kernel(const float* __restrict__ x,
                     const float* __restrict__ wq, const float* __restrict__ bq,
                     const float* __restrict__ wk, const float* __restrict__ bk,
                     const float* __restrict__ wv, const float* __restrict__ bv,
                     _Float16* __restrict__ Qh, _Float16* __restrict__ Kh,
                     _Float16* __restrict__ Vh)
{
    const int mBase = blockIdx.x * 64;     // token tile
    const int nt    = blockIdx.y;          // 0..23
    const int mat   = nt >> 3;             // 0=Q 1=K 2=V
    const int head  = nt & 7;
    const float* w    = (mat == 0) ? wq : (mat == 1) ? wk : wv;
    const float* bias = (mat == 0) ? bq : (mat == 1) ? bk : bv;
    _Float16*    dst  = (mat == 0) ? Qh : (mat == 1) ? Kh : Vh;

    __shared__ _Float16 Alds [64][40];   // [token][k]  64x32 + pad (80B pitch)
    __shared__ _Float16 BldsT[64][40];   // [n=d][k]    64x32 transposed + pad

    const int tid  = threadIdx.x;
    const int lane = tid & 31;
    const int wid  = tid >> 5;
    const int g    = lane >> 4;
    const int ln   = lane & 15;

    v8f acc[4] = {};

    for (int kc = 0; kc < DIM_; kc += 32) {
        __syncthreads();
        // stage A: 64x32 chunk of x, float4 loads -> v4h LDS stores
        #pragma unroll
        for (int it = 0; it < 4; ++it) {
            int j = tid + it * 128;          // 0..511 float4 chunks
            int r = j >> 3, c = (j & 7) * 4;
            float4 xv = *(const float4*)&x[(size_t)(mBase + r) * DIM_ + kc + c];
            v4h hv = { (_Float16)xv.x, (_Float16)xv.y, (_Float16)xv.z, (_Float16)xv.w };
            *(v4h*)&Alds[r][c] = hv;
        }
        // stage B transposed: w[kdim][n] -> BldsT[n][kdim]
        #pragma unroll
        for (int it = 0; it < 4; ++it) {
            int j = tid + it * 128;          // 0..511 float4 chunks of 32x64 tile
            int r = j >> 4, c = (j & 15) * 4; // r = kdim row, c = n col
            float4 wv4 = *(const float4*)&w[(size_t)(kc + r) * DIM_ + head * HD_ + c];
            BldsT[c + 0][r] = (_Float16)wv4.x;
            BldsT[c + 1][r] = (_Float16)wv4.y;
            BldsT[c + 2][r] = (_Float16)wv4.z;
            BldsT[c + 3][r] = (_Float16)wv4.w;
        }
        __syncthreads();

        v16h a;
        #pragma unroll
        for (int e = 0; e < 16; ++e) a[e] = Alds[wid * 16 + ln][kmap(e, g)];
        #pragma unroll
        for (int t = 0; t < 4; ++t) {
            v16h bf;
            #pragma unroll
            for (int e = 0; e < 16; ++e) bf[e] = BldsT[t * 16 + ln][kmap(e, g)];
            acc[t] = wmma_f32_16x16x32_f16(a, bf, acc[t]);
        }
    }

    // Epilogue: bias + RoPE in registers. C layout: row = r + 8*g (+16*wid),
    // col = t*16 + ln. Cols d and d+32 sit in tiles (t, t+2), same lane/VGPR r.
    float freqs[2], b_lo[2], b_hi[2];
    #pragma unroll
    for (int t = 0; t < 2; ++t) {
        int d0 = t * 16 + ln;
        freqs[t] = __expf((float)d0 * ROPE_C);   // 10000^(-d0/32)
        b_lo[t] = bias[head * HD_ + d0];
        b_hi[t] = bias[head * HD_ + d0 + 32];
    }
    #pragma unroll
    for (int r = 0; r < 8; ++r) {
        int row   = mBase + wid * 16 + g * 8 + r;   // global token 0..8191
        int b_idx = row >> 12;
        int spos  = row & 4095;
        #pragma unroll
        for (int t = 0; t < 2; ++t) {
            int   d0 = t * 16 + ln;
            float v0 = acc[t][r]     + b_lo[t];
            float v1 = acc[t + 2][r] + b_hi[t];
            float o0, o1;
            if (mat < 2) {                          // RoPE on Q,K only
                float theta = (float)spos * freqs[t];
                float sn = __sinf(theta), cs = __cosf(theta);
                o0 = v0 * cs - v1 * sn;
                o1 = v0 * sn + v1 * cs;
            } else {
                o0 = v0; o1 = v1;
            }
            size_t base = ((size_t)(b_idx * HEADS_ + head) * S_ + spos) * HD_;
            dst[base + d0]      = (_Float16)o0;
            dst[base + d0 + 32] = (_Float16)o1;
        }
    }
}

// ---------------------------------------------------------------------------
// Kernel 2: flash attention. One WG = (b,h, 128 q-rows); 4 waves x 32 q-rows.
// Q fragments loaded straight from global; K natural (already B-layout for
// Q*K^T); V staged transposed so P*V B-fragments gather contiguously.
// Every staged B-fragment feeds two WMMAs (two q-tiles per wave).
// ---------------------------------------------------------------------------
__global__ __launch_bounds__(128)
void flash_attn_kernel(const _Float16* __restrict__ Qh,
                       const _Float16* __restrict__ Kh,
                       const _Float16* __restrict__ Vh,
                       _Float16* __restrict__ O)
{
    const int qblk = blockIdx.x;       // 0..31 (128 q-rows each)
    const int bh   = blockIdx.y;       // 0..15
    const int bat  = bh >> 3;
    const int h    = bh & 7;

    const size_t headBase = (size_t)bh * S_ * HD_;
    const _Float16* Qg = Qh + headBase;
    const _Float16* Kg = Kh + headBase;
    const _Float16* Vg = Vh + headBase;

    __shared__ _Float16 Klds [64][72];      // [key][d]  (== B layout for QK^T)
    __shared__ _Float16 VldsT[64][72];      // [d][key]  (B layout for P*V)
    __shared__ _Float16 Plds [4][32][72];   // per-wave P staging

    const int tid  = threadIdx.x;
    const int lane = tid & 31;
    const int wid  = tid >> 5;
    const int g    = lane >> 4;
    const int ln   = lane & 15;

    // Q fragments: 2 q-tiles x 2 k-chunks, straight from global (16B runs).
    v16h a_q[2][2];
    #pragma unroll
    for (int qt = 0; qt < 2; ++qt) {
        int qrow = qblk * 128 + wid * 32 + qt * 16 + ln;
        const _Float16* qp = Qg + (size_t)qrow * HD_;
        #pragma unroll
        for (int kk = 0; kk < 2; ++kk)
            a_q[qt][kk] = load_frag_global(qp, kk, g);
    }

    v8f o_acc[2][4] = {};
    float m_i[2][8], l_i[2][8];
    #pragma unroll
    for (int qt = 0; qt < 2; ++qt)
        #pragma unroll
        for (int r = 0; r < 8; ++r) { m_i[qt][r] = -3.0e38f; l_i[qt][r] = 0.0f; }

    const float scale = 0.125f;        // 1/sqrt(64)

    for (int kb = 0; kb < S_ / 64; ++kb) {
        __syncthreads();
        // stage K natural + V transposed; v4h global loads
        #pragma unroll
        for (int it = 0; it < 8; ++it) {
            int j = tid + it * 128;          // 0..1023 chunks of 4 halves
            int r = j >> 4, c = (j & 15) * 4;
            v4h kv = *(const v4h*)(Kg + (size_t)(kb * 64 + r) * HD_ + c);
            v4h vv = *(const v4h*)(Vg + (size_t)(kb * 64 + r) * HD_ + c);
            *(v4h*)&Klds[r][c] = kv;
            VldsT[c + 0][r] = vv[0];
            VldsT[c + 1][r] = vv[1];
            VldsT[c + 2][r] = vv[2];
            VldsT[c + 3][r] = vv[3];
        }
        if (kb + 1 < S_ / 64) {        // next K/V tile -> global_prefetch_b8
            __builtin_prefetch(&Kg[(size_t)((kb + 1) * 64 + (tid >> 1)) * HD_], 0, 0);
            __builtin_prefetch(&Vg[(size_t)((kb + 1) * 64 + (tid >> 1)) * HD_], 0, 0);
        }
        __syncthreads();

        // S = Q * K^T : each K fragment feeds both q-tiles.
        v8f s_acc[2][4] = {};
        #pragma unroll
        for (int kk = 0; kk < 2; ++kk) {
            #pragma unroll
            for (int t = 0; t < 4; ++t) {
                v16h bf;
                #pragma unroll
                for (int e = 0; e < 16; ++e)
                    bf[e] = Klds[t * 16 + ln][kk * 32 + kmap(e, g)];
                s_acc[0][t] = wmma_f32_16x16x32_f16(a_q[0][kk], bf, s_acc[0][t]);
                s_acc[1][t] = wmma_f32_16x16x32_f16(a_q[1][kk], bf, s_acc[1][t]);
            }
        }

        // --- online softmax (per q-tile) ---
        #pragma unroll
        for (int qt = 0; qt < 2; ++qt) {
            float m_new[8];
            #pragma unroll
            for (int r = 0; r < 8; ++r) {
                float mx = -3.0e38f;
                #pragma unroll
                for (int t = 0; t < 4; ++t) mx = fmaxf(mx, s_acc[qt][t][r]);
                #pragma unroll
                for (int m = 1; m < 16; m <<= 1) mx = fmaxf(mx, __shfl_xor(mx, m, 32));
                m_new[r] = fmaxf(m_i[qt][r], mx * scale);
            }
            #pragma unroll
            for (int r = 0; r < 8; ++r) {
                float corr = __expf(m_i[qt][r] - m_new[r]);
                l_i[qt][r] *= corr;
                #pragma unroll
                for (int t = 0; t < 4; ++t) o_acc[qt][t][r] *= corr;
                m_i[qt][r] = m_new[r];
            }
            #pragma unroll
            for (int t = 0; t < 4; ++t) {
                #pragma unroll
                for (int r = 0; r < 8; ++r) {
                    float p = __expf(s_acc[qt][t][r] * scale - m_i[qt][r]);
                    s_acc[qt][t][r] = p;
                    Plds[wid][qt * 16 + g * 8 + r][t * 16 + ln] = (_Float16)p;
                }
            }
            #pragma unroll
            for (int r = 0; r < 8; ++r) {
                float sum = 0.0f;
                #pragma unroll
                for (int t = 0; t < 4; ++t) sum += s_acc[qt][t][r];
                #pragma unroll
                for (int m = 1; m < 16; m <<= 1) sum += __shfl_xor(sum, m, 32);
                l_i[qt][r] += sum;
            }
        }
        __syncthreads();   // P cross-lane LDS RAW

        // O += P * V : each V fragment feeds both q-tiles.
        v16h a_p[2][2];
        #pragma unroll
        for (int qt = 0; qt < 2; ++qt)
            #pragma unroll
            for (int kk = 0; kk < 2; ++kk)
                #pragma unroll
                for (int e = 0; e < 16; ++e)
                    a_p[qt][kk][e] = Plds[wid][qt * 16 + ln][kk * 32 + kmap(e, g)];
        #pragma unroll
        for (int t = 0; t < 4; ++t) {
            #pragma unroll
            for (int kk = 0; kk < 2; ++kk) {
                v16h bf;
                #pragma unroll
                for (int e = 0; e < 16; ++e)
                    bf[e] = VldsT[t * 16 + ln][kk * 32 + kmap(e, g)];
                o_acc[0][t] = wmma_f32_16x16x32_f16(a_p[0][kk], bf, o_acc[0][t]);
                o_acc[1][t] = wmma_f32_16x16x32_f16(a_p[1][kk], bf, o_acc[1][t]);
            }
        }
    }

    // Normalize and store O token-major [b][s][h*64+d].
    #pragma unroll
    for (int qt = 0; qt < 2; ++qt) {
        #pragma unroll
        for (int r = 0; r < 8; ++r) {
            float inv  = 1.0f / l_i[qt][r];
            int   srow = qblk * 128 + wid * 32 + qt * 16 + g * 8 + r;
            size_t ob  = ((size_t)bat * S_ + srow) * DIM_ + h * HD_;
            #pragma unroll
            for (int t = 0; t < 4; ++t)
                O[ob + t * 16 + ln] = (_Float16)(o_acc[qt][t][r] * inv);
        }
    }
}

// ---------------------------------------------------------------------------
// Kernel 3: output projection O(8192x512,f16) x Wo(512x512,f32) + bias -> f32
// ---------------------------------------------------------------------------
__global__ __launch_bounds__(128)
void out_proj_kernel(const _Float16* __restrict__ O,
                     const float* __restrict__ wo, const float* __restrict__ wo_b,
                     float* __restrict__ out)
{
    const int mBase = blockIdx.x * 64;
    const int nBase = blockIdx.y * 64;

    __shared__ _Float16 Alds [64][40];
    __shared__ _Float16 BldsT[64][40];   // [n][k] transposed

    const int tid  = threadIdx.x;
    const int lane = tid & 31;
    const int wid  = tid >> 5;
    const int g    = lane >> 4;
    const int ln   = lane & 15;

    v8f acc[4] = {};

    for (int kc = 0; kc < DIM_; kc += 32) {
        __syncthreads();
        #pragma unroll
        for (int it = 0; it < 4; ++it) {
            int j = tid + it * 128;          // 0..511 chunks of 4 halves
            int r = j >> 3, c = (j & 7) * 4;
            *(v4h*)&Alds[r][c] = *(const v4h*)&O[(size_t)(mBase + r) * DIM_ + kc + c];
        }
        #pragma unroll
        for (int it = 0; it < 4; ++it) {
            int j = tid + it * 128;
            int r = j >> 4, c = (j & 15) * 4;
            float4 wv4 = *(const float4*)&wo[(size_t)(kc + r) * DIM_ + nBase + c];
            BldsT[c + 0][r] = (_Float16)wv4.x;
            BldsT[c + 1][r] = (_Float16)wv4.y;
            BldsT[c + 2][r] = (_Float16)wv4.z;
            BldsT[c + 3][r] = (_Float16)wv4.w;
        }
        __syncthreads();

        v16h a;
        #pragma unroll
        for (int e = 0; e < 16; ++e) a[e] = Alds[wid * 16 + ln][kmap(e, g)];
        #pragma unroll
        for (int t = 0; t < 4; ++t) {
            v16h bf;
            #pragma unroll
            for (int e = 0; e < 16; ++e) bf[e] = BldsT[t * 16 + ln][kmap(e, g)];
            acc[t] = wmma_f32_16x16x32_f16(a, bf, acc[t]);
        }
    }

    #pragma unroll
    for (int r = 0; r < 8; ++r) {
        int row = mBase + wid * 16 + g * 8 + r;
        #pragma unroll
        for (int t = 0; t < 4; ++t) {
            int col = nBase + t * 16 + ln;
            out[(size_t)row * DIM_ + col] = acc[t][r] + wo_b[col];
        }
    }
}

// ---------------------------------------------------------------------------
extern "C" void kernel_launch(void* const* d_in, const int* in_sizes, int n_in,
                              void* d_out, int out_size, void* d_ws, size_t ws_size,
                              hipStream_t stream) {
    (void)in_sizes; (void)n_in; (void)out_size; (void)ws_size;

    const float* x   = (const float*)d_in[0];
    const float* wq  = (const float*)d_in[1];
    const float* bq  = (const float*)d_in[2];
    const float* wk  = (const float*)d_in[3];
    const float* bk  = (const float*)d_in[4];
    const float* wv  = (const float*)d_in[5];
    const float* bv  = (const float*)d_in[6];
    const float* wo  = (const float*)d_in[7];
    const float* wob = (const float*)d_in[8];
    float* out = (float*)d_out;

    const size_t elems = (size_t)ROWS_ * DIM_;           // 4M elements / tensor
    _Float16* Qh = (_Float16*)d_ws;
    _Float16* Kh = Qh + elems;
    _Float16* Vh = Kh + elems;
    _Float16* Oh = Vh + elems;

    qkv_rope_kernel<<<dim3(ROWS_ / 64, 24), 128, 0, stream>>>(
        x, wq, bq, wk, bk, wv, bv, Qh, Kh, Vh);
    flash_attn_kernel<<<dim3(S_ / 128, B_ * HEADS_), 128, 0, stream>>>(
        Qh, Kh, Vh, Oh);
    out_proj_kernel<<<dim3(ROWS_ / 64, DIM_ / 64), 128, 0, stream>>>(
        Oh, wo, wob, out);
}
